// LCGraphNet_11587821764949
// MI455X (gfx1250) — matrix operand order repaired and checked by reading, not compiled, not run on hardware
//
#include <hip/hip_runtime.h>
#include <hip/hip_bf16.h>
#include <math.h>

typedef __attribute__((ext_vector_type(16))) _Float16 v16h;
typedef __attribute__((ext_vector_type(8)))  _Float16 v8h;
typedef __attribute__((ext_vector_type(4)))  _Float16 v4h;
typedef __attribute__((ext_vector_type(8)))  float    v8f;

#define THREADS 256
#define WAVES   8
#define MTILE   32   // edges (or rows) per workgroup: two 16-row M-subtiles

// Build a 16-half WMMA fragment from two contiguous 8-half (16B) chunks.
__device__ __forceinline__ v16h make_frag(const _Float16* p0, const _Float16* p1) {
    v8h lo = *(const v8h*)p0;
    v8h hi = *(const v8h*)p1;
    v16h r;
#pragma unroll
    for (int i = 0; i < 8; ++i) { r[i] = lo[i]; r[8 + i] = hi[i]; }
    return r;
}

__device__ __forceinline__ v4h cvt4(float a, float b, float c, float d) {
    v4h r; r[0] = (_Float16)a; r[1] = (_Float16)b; r[2] = (_Float16)c; r[3] = (_Float16)d;
    return r;
}

// ---------------------------------------------------------------------------
// EdgeConv layer: msg = ReLU([x_i, x_j - x_i] @ W + b); agg = max over dst.
// agg must be pre-zeroed; ReLU >= 0 so float-max == u32-max on bit patterns.
// Wt is f16, transposed: [H][2D] (row = output column, contiguous K).
// Each wave owns one 16-wide N-tile and both 16-edge M-subtiles, so every
// B fragment is loaded once and feeds two independent WMMAs.
// ---------------------------------------------------------------------------
template <int D, int H>
__global__ __launch_bounds__(THREADS)
void edgeconv_kernel(const float* __restrict__ X,      // [N][D]
                     const int*   __restrict__ ei,     // [2][E] (src, dst)
                     const _Float16* __restrict__ Wt,  // [H][2D] f16
                     const float* __restrict__ bias,   // [H]
                     float* __restrict__ agg,          // [N][H], zero-initialized
                     int nEdges)
{
    constexpr int K2 = 2 * D;
    constexpr int NT = H / 16;
    constexpr int DV = D / 4;
    __shared__ __align__(16) _Float16 feat[MTILE][K2];
    __shared__ __align__(16) int sSrc[MTILE];
    __shared__ __align__(16) int sDst[MTILE];

    const int tid = threadIdx.x;
    const int e0  = blockIdx.x * MTILE;
    const bool full = (e0 + MTILE) <= nEdges;

    if (tid < MTILE) {
        int e = e0 + tid;
        if (e >= nEdges) e = nEdges - 1;
        sSrc[tid] = ei[e];
        sDst[tid] = ei[nEdges + e];
    }
    __syncthreads();

    // Gather + f32->f16 convert edge features into LDS: [x_i | x_j - x_i]
    const float4* X4 = (const float4*)X;
    for (int i = tid; i < MTILE * DV; i += THREADS) {
        const int r = i / DV, f = i - r * DV;
        const float4 xi = X4[(size_t)(unsigned)sDst[r] * DV + f];
        const float4 xj = X4[(size_t)(unsigned)sSrc[r] * DV + f];
        *(v4h*)&feat[r][4 * f]     = cvt4(xi.x, xi.y, xi.z, xi.w);
        *(v4h*)&feat[r][D + 4 * f] = cvt4(xj.x - xi.x, xj.y - xi.y, xj.z - xi.z, xj.w - xi.w);
    }
    __syncthreads();

    const int wave   = tid >> 5;
    const int lane   = tid & 31;
    const int nlo    = lane & 15;
    const int hiHalf = lane >> 4;           // 0: lanes 0-15, 1: lanes 16-31
    unsigned* aggU   = (unsigned*)agg;

    for (int nt = wave; nt < NT; nt += WAVES) {
        const int n0 = nt * 16;
        const _Float16* wrow = &Wt[(size_t)(n0 + nlo) * K2 + hiHalf * 8];
        v8f acc0 = {}, acc1 = {};
#pragma unroll
        for (int kk = 0; kk < K2; kk += 32) {
            // B (32x16 f16): col n0+nlo, halves K = kk+8*hiHalf+[0..7] and +16
            v16h b = make_frag(wrow + kk, wrow + kk + 16);
            // A (16x32 f16): row m, halves K = kk+8*hiHalf+[0..7] and +16
            const _Float16* ap0 = &feat[nlo][kk + hiHalf * 8];
            const _Float16* ap1 = &feat[16 + nlo][kk + hiHalf * 8];
            v16h a0 = make_frag(ap0, ap0 + 16);
            v16h a1 = make_frag(ap1, ap1 + 16);
            acc0 = __builtin_amdgcn_wmma_f32_16x16x32_f16(
                false, a0, false, b, (short)0, acc0, false, false);
            acc1 = __builtin_amdgcn_wmma_f32_16x16x32_f16(
                false, a1, false, b, (short)0, acc1, false, false);
        }

        // Epilogue: bias + ReLU + atomic max scatter to agg[dst]
        const int   n    = n0 + nlo;
        const float bn   = bias[n];
        const int   mOff = hiHalf * 8;       // C/D layout: VGPR r -> M = mOff + r

#pragma unroll
        for (int half = 0; half < 2; ++half) {
            const int mB = half * 16 + mOff;
            const int4 dlo = *(const int4*)&sDst[mB];
            const int4 dhi = *(const int4*)&sDst[mB + 4];
            const int dsts[8] = {dlo.x, dlo.y, dlo.z, dlo.w, dhi.x, dhi.y, dhi.z, dhi.w};
            float vals[8];
#pragma unroll
            for (int r = 0; r < 8; ++r) {
                float v = (half ? acc1[r] : acc0[r]) + bn;
                vals[r] = v > 0.f ? v : 0.f;
            }
            if (full) {
#pragma unroll
                for (int r = 0; r < 8; ++r)
                    atomicMax(aggU + (size_t)(unsigned)dsts[r] * H + n,
                              __float_as_uint(vals[r]));
            } else {
#pragma unroll
                for (int r = 0; r < 8; ++r)
                    if (e0 + mB + r < nEdges)
                        atomicMax(aggU + (size_t)(unsigned)dsts[r] * H + n,
                                  __float_as_uint(vals[r]));
            }
        }
    }
}

// ---------------------------------------------------------------------------
// Dense FC layer with ReLU: Y = ReLU(Xin @ W + b), WMMA-tiled like above.
// ---------------------------------------------------------------------------
template <int DIN, int H>
__global__ __launch_bounds__(THREADS)
void fc_kernel(const float* __restrict__ Xin,       // [N][DIN]
               const _Float16* __restrict__ Wt,     // [H][DIN] f16 transposed
               const float* __restrict__ bias,      // [H]
               float* __restrict__ Y,               // [N][H]
               int nRows)
{
    constexpr int NT = H / 16;
    constexpr int DV = DIN / 4;
    __shared__ __align__(16) _Float16 feat[MTILE][DIN];
    const int tid = threadIdx.x;
    const int r0  = blockIdx.x * MTILE;
    const bool full = (r0 + MTILE) <= nRows;

    const float4* X4 = (const float4*)Xin;
    for (int i = tid; i < MTILE * DV; i += THREADS) {
        const int r = i / DV, f = i - r * DV;
        int row = r0 + r;
        if (row >= nRows) row = nRows - 1;
        const float4 x = X4[(size_t)(unsigned)row * DV + f];
        *(v4h*)&feat[r][4 * f] = cvt4(x.x, x.y, x.z, x.w);
    }
    __syncthreads();

    const int wave   = tid >> 5;
    const int lane   = tid & 31;
    const int nlo    = lane & 15;
    const int hiHalf = lane >> 4;

    for (int nt = wave; nt < NT; nt += WAVES) {
        const int n0 = nt * 16;
        const _Float16* wrow = &Wt[(size_t)(n0 + nlo) * DIN + hiHalf * 8];
        v8f acc0 = {}, acc1 = {};
#pragma unroll
        for (int kk = 0; kk < DIN; kk += 32) {
            v16h b = make_frag(wrow + kk, wrow + kk + 16);
            const _Float16* ap0 = &feat[nlo][kk + hiHalf * 8];
            const _Float16* ap1 = &feat[16 + nlo][kk + hiHalf * 8];
            v16h a0 = make_frag(ap0, ap0 + 16);
            v16h a1 = make_frag(ap1, ap1 + 16);
            acc0 = __builtin_amdgcn_wmma_f32_16x16x32_f16(
                false, a0, false, b, (short)0, acc0, false, false);
            acc1 = __builtin_amdgcn_wmma_f32_16x16x32_f16(
                false, a1, false, b, (short)0, acc1, false, false);
        }

        const int   n    = n0 + nlo;
        const float bn   = bias[n];
        const int   mOff = hiHalf * 8;
#pragma unroll
        for (int half = 0; half < 2; ++half) {
            const int mB = half * 16 + mOff;
#pragma unroll
            for (int r = 0; r < 8; ++r) {
                float v = (half ? acc1[r] : acc0[r]) + bn;
                v = v > 0.f ? v : 0.f;
                const int row = r0 + mB + r;
                if (full || row < nRows)
                    Y[(size_t)(unsigned)row * H + n] = v;
            }
        }
    }
}

// Final head: out[i] = sigmoid(dot(fc1[i,:256], Wf2) + bf2)
__global__ void head_kernel(const float* __restrict__ fc1,
                            const float* __restrict__ w,
                            const float* __restrict__ b,
                            float* __restrict__ out, int n)
{
    const int i = blockIdx.x * blockDim.x + threadIdx.x;
    if (i >= n) return;
    const float4* a  = (const float4*)(fc1 + (size_t)i * 256);
    const float4* ww = (const float4*)w;
    float s = 0.f;
#pragma unroll 8
    for (int j = 0; j < 64; ++j) {
        float4 x = a[j], y = ww[j];
        s += x.x * y.x + x.y * y.y + x.z * y.z + x.w * y.w;
    }
    s += b[0];
    out[i] = 1.0f / (1.0f + expf(-s));
}

// W [K][H] f32 -> Wt [H][K] f16 (transposed so K is contiguous per out-col)
__global__ void convert_wt_kernel(const float* __restrict__ W,
                                  _Float16* __restrict__ Wt, int K, int H)
{
    const int i = blockIdx.x * blockDim.x + threadIdx.x;
    if (i < K * H) {
        const int k = i / H, n = i - k * H;
        Wt[(size_t)n * K + k] = (_Float16)W[i];
    }
}

__global__ void zero_kernel(float* __restrict__ p, size_t n)
{
    size_t i = (size_t)blockIdx.x * blockDim.x + threadIdx.x;
    const size_t stride = (size_t)gridDim.x * blockDim.x;
    for (; i < n; i += stride) p[i] = 0.f;
}

extern "C" void kernel_launch(void* const* d_in, const int* in_sizes, int n_in,
                              void* d_out, int out_size, void* d_ws, size_t ws_size,
                              hipStream_t stream)
{
    (void)n_in; (void)out_size; (void)ws_size;

    const float* X   = (const float*)d_in[0];
    const int*   ei  = (const int*)d_in[1];
    const float* W1  = (const float*)d_in[2];  const float* b1  = (const float*)d_in[3];
    const float* W2  = (const float*)d_in[4];  const float* b2  = (const float*)d_in[5];
    const float* W3  = (const float*)d_in[6];  const float* b3  = (const float*)d_in[7];
    const float* Wf1 = (const float*)d_in[8];  const float* bf1 = (const float*)d_in[9];
    const float* Wf2 = (const float*)d_in[10]; const float* bf2 = (const float*)d_in[11];

    const int N = in_sizes[0] / 16;   // 50000
    const int E = in_sizes[1] / 2;    // 800000

    // Workspace layout (floats):
    //   regionA [N*256]: agg1 (N*64) | agg2 (N*128) ... later reused as fc1 (N*256)
    //   agg3    [N*256]
    //   then f16 transposed weights
    float* ws    = (float*)d_ws;
    float* agg1  = ws;
    float* agg2  = ws + (size_t)N * 64;
    float* fc1   = ws;                         // aliases agg1|agg2 (dead by then)
    float* agg3  = ws + (size_t)N * 256;
    _Float16* W1t  = (_Float16*)(ws + (size_t)N * 512);
    _Float16* W2t  = W1t + 32 * 64;
    _Float16* W3t  = W2t + 128 * 128;
    _Float16* Wf1t = W3t + 256 * 256;

    // 1) Zero all max-aggregation buffers (every call: atomics accumulate).
    zero_kernel<<<2048, THREADS, 0, stream>>>(ws, (size_t)N * 512);

    // 2) Convert weights to f16, transposed [H][K].
    convert_wt_kernel<<<(32 * 64  + 255) / 256, 256, 0, stream>>>(W1,  W1t,  32,  64);
    convert_wt_kernel<<<(128*128 + 255) / 256, 256, 0, stream>>>(W2,  W2t,  128, 128);
    convert_wt_kernel<<<(256*256 + 255) / 256, 256, 0, stream>>>(W3,  W3t,  256, 256);
    convert_wt_kernel<<<(256*256 + 255) / 256, 256, 0, stream>>>(Wf1, Wf1t, 256, 256);

    const int edgeBlocks = (E + MTILE - 1) / MTILE;

    // 3) Three EdgeConv layers (WMMA GEMM per edge tile + atomic-max scatter).
    edgeconv_kernel<16, 64 ><<<edgeBlocks, THREADS, 0, stream>>>(X,    ei, W1t, b1, agg1, E);
    edgeconv_kernel<64, 128><<<edgeBlocks, THREADS, 0, stream>>>(agg1, ei, W2t, b2, agg2, E);
    edgeconv_kernel<128,256><<<edgeBlocks, THREADS, 0, stream>>>(agg2, ei, W3t, b3, agg3, E);

    // 4) MLP head.
    fc_kernel<256, 256><<<(N + MTILE - 1) / MTILE, THREADS, 0, stream>>>(agg3, Wf1t, bf1, fc1, N);
    head_kernel<<<(N + 255) / 256, 256, 0, stream>>>(fc1, Wf2, bf2, (float*)d_out, N);
}
